// GAT_75634374082904
// MI455X (gfx1250) — matrix operand order repaired and checked by reading, not compiled, not run on hardware
//
#include <hip/hip_runtime.h>
#include <hip/hip_bf16.h>

// ---------------------------------------------------------------------------
// GraphSAGE-style layer: out = h@W1 + b1 + mean_neighbor(h)@W2 + b2
//   Phase 0 (prep):  zero out-buffer (used as neigh_sum accumulator) + deg,
//                    split W1/W2 into bf16 hi/lo, TRANSPOSED to [N,K],
//                    bias = b1+b2.
//   Phase 1 (edges): 1 wave per edge; float4 gather of h[src], 4x
//                    global_atomic_add_f32 into neigh_sum (=d_out), deg++.
//   Phase 2 (gemm):  v_wmma_f32_16x16x32_bf16, bf16 hi/lo split (3 products)
//                    for ~f32 accuracy; per-row 1/max(deg,1) folded into the
//                    A2 conversion; in-place overwrite of d_out.
// ---------------------------------------------------------------------------

typedef __bf16 bf16_t;
typedef __bf16 bf16x8  __attribute__((ext_vector_type(8)));
typedef __bf16 bf16x16 __attribute__((ext_vector_type(16)));
typedef float  f32x8   __attribute__((ext_vector_type(8)));

#define D 128          // D_IN == D_OUT == 128 (reference)

// workspace byte offsets (all 512-aligned); total ~332 KB
#define WS_DEG   0u
#define WS_W1H   204800u
#define WS_W1L   (WS_W1H + 32768u)
#define WS_W2H   (WS_W1L + 32768u)
#define WS_W2L   (WS_W2H + 32768u)
#define WS_BIAS  (WS_W2L + 32768u)

// ---------------------------------------------------------------------------
// Phase 0: zero accumulators, convert weights (bf16 hi/lo, transposed), bias
// ---------------------------------------------------------------------------
__global__ void gnn_prep_kernel(const float* __restrict__ W1, const float* __restrict__ b1,
                                const float* __restrict__ W2, const float* __restrict__ b2,
                                float* __restrict__ nsum /* == d_out */,
                                float* __restrict__ deg,
                                bf16_t* __restrict__ w1h, bf16_t* __restrict__ w1l,
                                bf16_t* __restrict__ w2h, bf16_t* __restrict__ w2l,
                                float* __restrict__ bias, int n_nodes)
{
    int idx = blockIdx.x * blockDim.x + threadIdx.x;
    int total = n_nodes * D;
    if (idx < total)  nsum[idx] = 0.0f;
    if (idx < n_nodes) deg[idx] = 0.0f;
    if (idx < D * D) {
        int k = idx / D, n = idx % D;            // W is [K,N] row-major
        float x1 = W1[idx];
        bf16_t h1 = (bf16_t)x1;
        bf16_t l1 = (bf16_t)(x1 - (float)h1);
        w1h[n * D + k] = h1;                     // store transposed [N,K]
        w1l[n * D + k] = l1;
        float x2 = W2[idx];
        bf16_t h2 = (bf16_t)x2;
        bf16_t l2 = (bf16_t)(x2 - (float)h2);
        w2h[n * D + k] = h2;
        w2l[n * D + k] = l2;
    }
    if (idx < D) bias[idx] = b1[idx] + b2[idx];
}

// ---------------------------------------------------------------------------
// Phase 1: edge scatter-add.  One wave32 per edge, float4 per lane.
// h fits in L2 (25.6 MB << 192 MB) -> random row gathers hit L2.
// ---------------------------------------------------------------------------
__global__ void gnn_edge_kernel(const float* __restrict__ h,
                                const int* __restrict__ src,
                                const int* __restrict__ dst,
                                float* __restrict__ nsum,
                                float* __restrict__ deg, int n_edges)
{
    int wave = (blockIdx.x * blockDim.x + threadIdx.x) >> 5;   // edge id
    int lane = threadIdx.x & 31;
    if (wave >= n_edges) return;
    int s = src[wave];
    int d = dst[wave];
    const float4* hp = (const float4*)(h + (size_t)s * D);
    float4 v = hp[lane];                                       // global_load_b128
    float* np = nsum + (size_t)d * D + lane * 4;
    atomicAdd(np + 0, v.x);
    atomicAdd(np + 1, v.y);
    atomicAdd(np + 2, v.z);
    atomicAdd(np + 3, v.w);
    if (lane == 0) atomicAdd(deg + d, 1.0f);
}

// ---------------------------------------------------------------------------
// Phase 2: WMMA GEMM
// ---------------------------------------------------------------------------
__device__ __forceinline__ f32x8 wmma_bf16(bf16x16 a, bf16x16 b, f32x8 c) {
    return __builtin_amdgcn_wmma_f32_16x16x32_bf16(
        /*neg_a=*/false, a, /*neg_b=*/false, b,
        /*c_mod=*/(short)0, c, /*reuse_a=*/false, /*reuse_b=*/false);
}

struct AFrag { bf16x16 hi, lo; };

// A 16x32 bf16 fragment in this code's contiguous k order (B matches): lane L holds row M=L%16;
//   VGPR0..7 = K = 16*(L/16)+{0..15} (packed pairs).
// Source is f32 row-major -> 4x float4 loads, convert with hi/lo split,
// with per-row scale folded in (scale=1 for h, 1/max(deg,1) for neigh_sum).
__device__ __forceinline__ AFrag load_a_f32(const float* __restrict__ row,
                                            int kb, int koffA, float scale)
{
    const float4* p0 = (const float4*)(row + kb + koffA);
    const float4* p1 = (const float4*)(row + kb + 8 + koffA);
    float4 f0 = p0[0], f1 = p0[1], f2 = p1[0], f3 = p1[1];
    float f[16] = { f0.x, f0.y, f0.z, f0.w,  f1.x, f1.y, f1.z, f1.w,
                    f2.x, f2.y, f2.z, f2.w,  f3.x, f3.y, f3.z, f3.w };
    AFrag r;
#pragma unroll
    for (int j = 0; j < 16; ++j) {
        float x = f[j] * scale;
        bf16_t hi = (bf16_t)x;
        r.hi[j] = hi;
        r.lo[j] = (bf16_t)(x - (float)hi);
    }
    return r;
}

// B 32x16 bf16 fragment in the same contiguous k order as A: lane L holds col N=L%16;
//   lanes 0-15: K=kb+0..15, lanes 16-31: K=kb+16..31 (packed pairs across
//   VGPR0..7).  Weights stored transposed [N,K] -> 16 contiguous bf16 = 2xb128.
__device__ __forceinline__ bf16x16 load_b(const bf16_t* __restrict__ wt,
                                          int nrow, int kb, int koffB)
{
    const bf16x8* p = (const bf16x8*)(wt + nrow * D + kb + koffB);
    bf16x8 lo = p[0], hi = p[1];
    return __builtin_shufflevector(lo, hi, 0,1,2,3,4,5,6,7,8,9,10,11,12,13,14,15);
}

__global__ void __launch_bounds__(256)
gnn_gemm_kernel(const float* __restrict__ h,
                float* __restrict__ out /* also neigh_sum input, in-place */,
                const float* __restrict__ deg,
                const bf16_t* __restrict__ w1h, const bf16_t* __restrict__ w1l,
                const bf16_t* __restrict__ w2h, const bf16_t* __restrict__ w2l,
                const float* __restrict__ bias, int n_nodes)
{
    int wave = threadIdx.x >> 5;
    int lane = threadIdx.x & 31;
    int m_tiles = (n_nodes + 15) >> 4;
    int tile = blockIdx.x * 8 + wave;
    if (tile >= m_tiles) return;          // uniform per wave: EXEC stays all-1s

    int r0    = tile * 16;
    int lrow  = r0 + (lane & 15);
    int rowc  = lrow < n_nodes ? lrow : (n_nodes - 1);  // clamp for tail loads
    int koffA = (lane >> 4) * 16;         // A-fragment K sub-offset
    int koffB = (lane >> 4) * 16;         // B-fragment K sub-offset

    float s = 1.0f / fmaxf(deg[rowc], 1.0f);   // mean normalization (DGL conv.)

    const float* arow = h   + (size_t)rowc * D;
    const float* nrow = out + (size_t)rowc * D;    // neigh_sum lives in d_out

    // init accumulators with fused bias (depends only on column)
    f32x8 acc[8];
#pragma unroll
    for (int n = 0; n < 8; ++n) {
        float b = bias[n * 16 + (lane & 15)];
#pragma unroll
        for (int v = 0; v < 8; ++v) acc[n][v] = b;
    }

#pragma unroll
    for (int kk = 0; kk < 4; ++kk) {      // K = 128 in 4 steps of 32
        int kb = kk * 32;
        AFrag a1 = load_a_f32(arow, kb, koffA, 1.0f);  // h
        AFrag a2 = load_a_f32(nrow, kb, koffA, s);     // h_N = s * neigh_sum
#pragma unroll
        for (int n = 0; n < 8; ++n) {
            int nr = n * 16 + (lane & 15);
            bf16x16 b1h = load_b(w1h, nr, kb, koffB);
            bf16x16 b1l = load_b(w1l, nr, kb, koffB);
            bf16x16 b2h = load_b(w2h, nr, kb, koffB);
            bf16x16 b2l = load_b(w2l, nr, kb, koffB);
            // bf16 hi/lo split: A*B ~= Ah*Bh + Ah*Bl + Al*Bh  (~1e-4 rel err)
            acc[n] = wmma_bf16(a1.hi, b1h, acc[n]);
            acc[n] = wmma_bf16(a1.hi, b1l, acc[n]);
            acc[n] = wmma_bf16(a1.lo, b1h, acc[n]);
            acc[n] = wmma_bf16(a2.hi, b2h, acc[n]);
            acc[n] = wmma_bf16(a2.hi, b2l, acc[n]);
            acc[n] = wmma_bf16(a2.lo, b2h, acc[n]);
        }
    }

    // C/D layout: lanes 0-15 -> M=v, lanes 16-31 -> M=v+8; col = n*16 + lane%16
#pragma unroll
    for (int v = 0; v < 8; ++v) {
        int row = r0 + v + ((lane >> 4) << 3);
        if (row < n_nodes) {
            float* orow = out + (size_t)row * D + (lane & 15);
#pragma unroll
            for (int n = 0; n < 8; ++n) orow[n * 16] = acc[n][v];
        }
    }
}

// ---------------------------------------------------------------------------
extern "C" void kernel_launch(void* const* d_in, const int* in_sizes, int n_in,
                              void* d_out, int out_size, void* d_ws, size_t ws_size,
                              hipStream_t stream)
{
    const float* h   = (const float*)d_in[0];
    const int*   src = (const int*)  d_in[1];
    const int*   dst = (const int*)  d_in[2];
    const float* W1  = (const float*)d_in[3];
    const float* b1  = (const float*)d_in[4];
    const float* W2  = (const float*)d_in[5];
    const float* b2  = (const float*)d_in[6];

    int n_nodes = in_sizes[0] / D;
    int n_edges = in_sizes[1];

    float*  out  = (float*)d_out;
    char*   ws   = (char*)d_ws;
    float*  deg  = (float*)(ws + WS_DEG);
    bf16_t* w1h  = (bf16_t*)(ws + WS_W1H);
    bf16_t* w1l  = (bf16_t*)(ws + WS_W1L);
    bf16_t* w2h  = (bf16_t*)(ws + WS_W2H);
    bf16_t* w2l  = (bf16_t*)(ws + WS_W2L);
    float*  bias = (float*)(ws + WS_BIAS);

    // Phase 0: zero accumulators + convert weights + bias
    {
        int total  = n_nodes * D;
        int blocks = (total + 255) / 256;
        gnn_prep_kernel<<<blocks, 256, 0, stream>>>(W1, b1, W2, b2, out, deg,
                                                    w1h, w1l, w2h, w2l, bias,
                                                    n_nodes);
    }
    // Phase 1: edge scatter-add (1 wave per edge, 8 edges per 256-thread block)
    {
        int blocks = (n_edges + 7) / 8;
        gnn_edge_kernel<<<blocks, 256, 0, stream>>>(h, src, dst, out, deg,
                                                    n_edges);
    }
    // Phase 2: fused dual-GEMM with WMMA (8 waves = 8 M-tiles per block)
    {
        int m_tiles = (n_nodes + 15) / 16;
        int blocks  = (m_tiles + 7) / 8;
        gnn_gemm_kernel<<<blocks, 256, 0, stream>>>(h, out, deg,
                                                    w1h, w1l, w2h, w2l, bias,
                                                    n_nodes);
    }
}
